// MultiHeadAttention_23562190586285
// MI455X (gfx1250) — matrix operand order repaired and checked by reading, not compiled, not run on hardware
//
#include <hip/hip_runtime.h>

typedef unsigned short u16;
typedef __attribute__((ext_vector_type(16))) __bf16 v16bf;
typedef __attribute__((ext_vector_type(8)))  float  v8f;

union frag16 {
    v16bf v;
    u16   u[16];
    uint4 q[2];
};

#define SEQ 2048
#define DMODEL 1024
#define DK 64

__device__ __forceinline__ u16 f32_to_bf16(float f) {
    unsigned u = __builtin_bit_cast(unsigned, f);
    unsigned r = u + 0x7FFFu + ((u >> 16) & 1u);   // round-to-nearest-even
    return (u16)(r >> 16);
}

// ---------------------------------------------------------------------------
// fp32 -> bf16 elementwise convert (grid-stride)
// ---------------------------------------------------------------------------
__global__ void cvt_bf16(const float* __restrict__ src, u16* __restrict__ dst, int n) {
    int i = blockIdx.x * blockDim.x + threadIdx.x;
    int stride = gridDim.x * blockDim.x;
    for (; i < n; i += stride) dst[i] = f32_to_bf16(src[i]);
}

// ---------------------------------------------------------------------------
// C[m,n] = sum_k A[m,k] * W[n,k] + bias[n]   (torch "x @ W.T + b")
// A: bf16 [M,K] row-major, W: bf16 [N,K] row-major (W rows == WMMA B columns,
// so every fragment load is contiguous).
// Each wave owns a 16x64 strip. The k-loop is software-pipelined with two
// distinct register buffers (a0/b0[] vs a1/b1[]): the next 32-chunk's loads
// are issued before the current chunk's 4 WMMAs, so waits become partial
// instead of s_wait_loadcnt 0 per WMMA. Requires K % 64 == 0.
// mode 0: fp32 store to Cf, plain [M,N]
// mode 1: bf16 store to Cb as [B,H,S,64]   (Q / K layout)
// mode 2: bf16 store to Cb as [B,H,64,S]   (V transposed layout)
// ---------------------------------------------------------------------------
__global__ void __launch_bounds__(256) gemm_wmma_nt(
    const u16* __restrict__ A, const u16* __restrict__ W,
    const float* __restrict__ bias, float* __restrict__ Cf,
    u16* __restrict__ Cb, int M, int N, int K, int mode)
{
    int lane = threadIdx.x & 31;
    int wave = threadIdx.x >> 5;
    int strips_n = N >> 6;                       // 64-wide strips
    int strip = blockIdx.x * 8 + wave;
    if (strip >= (M >> 4) * strips_n) return;
    int tm = strip / strips_n;
    int sn = strip - tm * strips_n;

    int rowA = (tm << 4) + (lane & 15);          // A fragment: row m = lane%16
    int loA  = (lane >> 4) * 16;                 // A k-run start within chunk
    int loB  = (lane >> 4) * 16;                 // B k-run start within chunk

    const u16* Ap  = A + (size_t)rowA * K;
    const u16* Bp0 = W + (size_t)((sn << 6) + (lane & 15)) * K;
    const size_t BntStride = (size_t)16 * K;     // next 16-column subtile

    v8f acc[4] = {};
    frag16 a0, a1, b0[4], b1[4];

    auto loadA = [&](frag16& a, int kk) {
        a.q[0] = *(const uint4*)(Ap + kk + loA);        // k = lo .. lo+7
        a.q[1] = *(const uint4*)(Ap + kk + loA + 8);    // k = lo+8 .. lo+15
    };
    auto loadB = [&](frag16* b, int kk) {
#pragma unroll
        for (int nt = 0; nt < 4; ++nt) {
            const u16* bp = Bp0 + nt * BntStride + kk;
            b[nt].q[0] = *(const uint4*)(bp + loB);      // k = loB .. loB+7
            b[nt].q[1] = *(const uint4*)(bp + loB + 8);  // k = loB+8 .. loB+15
        }
    };
    auto mma4 = [&](frag16& a, frag16* b) {
#pragma unroll
        for (int nt = 0; nt < 4; ++nt)
            acc[nt] = __builtin_amdgcn_wmma_f32_16x16x32_bf16(
                          false, a.v, false, b[nt].v, (short)0, acc[nt], false, false);
    };

    loadA(a0, 0);
    loadB(b0, 0);
    int kb = 0;
    for (; kb + 64 < K; kb += 64) {
        loadA(a1, kb + 32);
        loadB(b1, kb + 32);
        // speculative prefetch ahead of the streams (global_prefetch_b8;
        // dropped silently past the end per ISA 10.5)
        __builtin_prefetch(Ap + kb + 512, 0, 1);
        __builtin_prefetch(Bp0 + kb + 512, 0, 1);
        mma4(a0, b0);
        loadA(a0, kb + 64);
        loadB(b0, kb + 64);
        mma4(a1, b1);
    }
    // epilogue: kb == K - 64
    loadA(a1, kb + 32);
    loadB(b1, kb + 32);
    mma4(a0, b0);
    mma4(a1, b1);

    int m0 = (tm << 4) + ((lane >> 4) << 3);
#pragma unroll
    for (int nt = 0; nt < 4; ++nt) {
        int n = (sn << 6) + (nt << 4) + (lane & 15);
        float bn = bias ? bias[n] : 0.0f;
#pragma unroll
        for (int r = 0; r < 8; ++r) {
            int m = m0 + r;
            float val = acc[nt][r] + bn;
            if (mode == 0) {
                Cf[(size_t)m * N + n] = val;
            } else {
                int b_ = m >> 11, s_ = m & 2047;    // M = B*S = 2*2048
                int h_ = n >> 6,  d_ = n & 63;      // N = H*DK = 16*64
                size_t idx = (mode == 1)
                    ? ((size_t)((b_ * 16 + h_) * SEQ + s_) * DK + d_)
                    : ((size_t)((b_ * 16 + h_) * DK + d_) * SEQ + s_);
                Cb[idx] = f32_to_bf16(val);
            }
        }
    }
}

// ---------------------------------------------------------------------------
// Fused attention: one workgroup (8 waves) per (b,h, 16-row q-block).
//  Phase A: logits tile L[16][2048] = Q_tile @ K^T / 8   via WMMA -> LDS
//           (128 KB tile: only possible with CDNA5's 320 KB/WGP LDS).
//           K-tile B fragments are ping-pong double-buffered across tiles.
//  Phase B: sparsemax per row; tau by bisection on f(t)=sum relu(z-t)-1
//           (register-resident row, 64 f32/lane, wave32 shfl reductions).
//           Streams the 512 MB attention tensor out coalesced (HBM floor).
//  Phase C: O_tile[16][64] = P @ V via WMMA (P from LDS, cvt to bf16 on the
//           fly; V fragments ping-pong buffered). Cross-wave ds_add_f32.
// ---------------------------------------------------------------------------
__global__ void __launch_bounds__(256) attn_sparsemax(
    const u16* __restrict__ Q, const u16* __restrict__ Kt,
    const u16* __restrict__ Vt, float* __restrict__ attnG,
    u16* __restrict__ Ob)
{
    __shared__ float Lt[16 * SEQ];     // 128 KB logits / P tile
    __shared__ float Osh[16 * DK];     // 4 KB  cross-wave O reduction

    int lane = threadIdx.x & 31;
    int wave = threadIdx.x >> 5;
    int blk  = blockIdx.x;
    int bh   = blk >> 7;               // / (SEQ/16 = 128)
    int q0   = (blk & 127) << 4;

    const u16* Qb = Q  + (size_t)bh * SEQ * DK;
    const u16* Kb = Kt + (size_t)bh * SEQ * DK;
    const u16* Vb = Vt + (size_t)bh * DK * SEQ;

    // ---- Phase A: logits -------------------------------------------------
    int lo  = (lane >> 4) * 16;
    int loB = (lane >> 4) * 16;
    const u16* qp = Qb + (size_t)(q0 + (lane & 15)) * DK;
    frag16 a0, a1;
    a0.q[0] = *(const uint4*)(qp + lo);
    a0.q[1] = *(const uint4*)(qp + lo + 8);
    a1.q[0] = *(const uint4*)(qp + 32 + lo);
    a1.q[1] = *(const uint4*)(qp + 32 + lo + 8);

    frag16 kf[2][2];                   // ping-pong K-tile fragments
    auto loadK = [&](frag16* bb, int t) {
        int n0 = (wave * 16 + t) * 16;
        const u16* kp = Kb + (size_t)(n0 + (lane & 15)) * DK;
        bb[0].q[0] = *(const uint4*)(kp + loB);
        bb[0].q[1] = *(const uint4*)(kp + loB + 8);
        bb[1].q[0] = *(const uint4*)(kp + 32 + loB);
        bb[1].q[1] = *(const uint4*)(kp + 32 + loB + 8);
        __builtin_prefetch(kp + 16 * DK, 0, 1);
    };

    loadK(kf[0], 0);
#pragma unroll
    for (int t = 0; t < 16; ++t) {
        int cur = t & 1;
        if (t + 1 < 16) loadK(kf[cur ^ 1], t + 1);
        v8f acc = {};
        acc = __builtin_amdgcn_wmma_f32_16x16x32_bf16(
                  false, a0.v, false, kf[cur][0].v, (short)0, acc, false, false);
        acc = __builtin_amdgcn_wmma_f32_16x16x32_bf16(
                  false, a1.v, false, kf[cur][1].v, (short)0, acc, false, false);
        int mr = (lane >> 4) << 3;
        int n  = (wave * 16 + t) * 16 + (lane & 15);
#pragma unroll
        for (int r = 0; r < 8; ++r)
            Lt[(mr + r) * SEQ + n] = acc[r] * 0.125f;   // 1/sqrt(64)
    }
    __syncthreads();

    // ---- Phase B: sparsemax ---------------------------------------------
    for (int rr = 0; rr < 2; ++rr) {
        int r = wave * 2 + rr;
        float z[64];
#pragma unroll
        for (int j = 0; j < 64; ++j) z[j] = Lt[r * SEQ + lane + 32 * j];

        float mx = -3.4e38f;
#pragma unroll
        for (int j = 0; j < 64; ++j) mx = fmaxf(mx, z[j]);
        for (int off = 16; off; off >>= 1) mx = fmaxf(mx, __shfl_xor(mx, off, 32));

        // f(tau) = sum relu(z - tau) - 1 is strictly decreasing;
        // f(mx-1) >= 0 and f(mx) = -1, so bisect on [mx-1, mx].
        float tlo = mx - 1.0f, thi = mx;
        for (int it = 0; it < 26; ++it) {
            float tau = 0.5f * (tlo + thi);
            float s = 0.0f;
#pragma unroll
            for (int j = 0; j < 64; ++j) s += fmaxf(z[j] - tau, 0.0f);
            for (int off = 16; off; off >>= 1) s += __shfl_xor(s, off, 32);
            if (s >= 1.0f) tlo = tau; else thi = tau;
        }
        float tau = 0.5f * (tlo + thi);

        float* arow = attnG + ((size_t)bh * SEQ + (q0 + r)) * SEQ;
#pragma unroll
        for (int j = 0; j < 64; ++j) {
            float p = fmaxf(z[j] - tau, 0.0f);
            arow[lane + 32 * j] = p;            // coalesced 512MB stream out
            Lt[r * SEQ + lane + 32 * j] = p;    // keep P for phase C
        }
    }
    __syncthreads();

    // ---- Phase C: O = P @ V ---------------------------------------------
    for (int i = threadIdx.x; i < 16 * DK; i += 256) Osh[i] = 0.0f;
    __syncthreads();

    v8f acc[4] = {};
    int mA = lane & 15;
    frag16 vf[2][4];                   // ping-pong V fragments
    auto loadV = [&](frag16* vv, int kb) {
#pragma unroll
        for (int nt = 0; nt < 4; ++nt) {
            const u16* vp = Vb + (size_t)(nt * 16 + (lane & 15)) * SEQ + kb + loB;
            vv[nt].q[0] = *(const uint4*)vp;
            vv[nt].q[1] = *(const uint4*)(vp + 8);
        }
    };

    loadV(vf[0], wave * 32);
#pragma unroll
    for (int i = 0; i < 8; ++i) {
        int kb = (wave + 8 * i) * 32;
        int cur = i & 1;
        if (i + 1 < 8) loadV(vf[cur ^ 1], kb + 8 * 32);
        frag16 a;
        const float* lp = &Lt[mA * SEQ + kb + lo];
#pragma unroll
        for (int j = 0; j < 8; ++j) a.u[j]     = f32_to_bf16(lp[j]);
#pragma unroll
        for (int j = 0; j < 8; ++j) a.u[8 + j] = f32_to_bf16(lp[8 + j]);
#pragma unroll
        for (int nt = 0; nt < 4; ++nt)
            acc[nt] = __builtin_amdgcn_wmma_f32_16x16x32_bf16(
                          false, a.v, false, vf[cur][nt].v, (short)0, acc[nt], false, false);
    }
    int mr = (lane >> 4) << 3;
#pragma unroll
    for (int nt = 0; nt < 4; ++nt) {
        int d = nt * 16 + (lane & 15);
#pragma unroll
        for (int r = 0; r < 8; ++r)
            atomicAdd(&Osh[(mr + r) * DK + d], acc[nt][r]);  // ds_add_f32
    }
    __syncthreads();

    int b_ = bh >> 4, h_ = bh & 15;
    for (int i = threadIdx.x; i < 16 * DK; i += 256) {
        int m = i >> 6, d = i & 63;
        size_t oi = ((size_t)(b_ * SEQ + q0 + m)) * DMODEL + h_ * DK + d;
        Ob[oi] = f32_to_bf16(Osh[i]);   // [B,S,D] layout for the final GEMM
    }
}

// ---------------------------------------------------------------------------
extern "C" void kernel_launch(void* const* d_in, const int* in_sizes, int n_in,
                              void* d_out, int out_size, void* d_ws, size_t ws_size,
                              hipStream_t stream) {
    const float* query = (const float*)d_in[0];
    const float* key   = (const float*)d_in[1];
    const float* value = (const float*)d_in[2];
    const float* Wq  = (const float*)d_in[3];
    const float* bq  = (const float*)d_in[4];
    const float* Wk  = (const float*)d_in[5];
    const float* bk  = (const float*)d_in[6];
    const float* Wv  = (const float*)d_in[7];
    const float* bv  = (const float*)d_in[8];
    const float* Wfc = (const float*)d_in[9];
    const float* bfc = (const float*)d_in[10];

    const size_t NX = (size_t)2 * SEQ * DMODEL;   // 4M activations
    const size_t NW = (size_t)DMODEL * DMODEL;    // 1M weights

    u16* ws   = (u16*)d_ws;
    u16* Xq   = ws;
    u16* Xk   = Xq  + NX;
    u16* Xv   = Xk  + NX;
    u16* Wqb  = Xv  + NX;
    u16* Wkb  = Wqb + NW;
    u16* Wvb  = Wkb + NW;
    u16* Wfcb = Wvb + NW;
    u16* Qb   = Wfcb + NW;
    u16* Kbf  = Qb  + NX;
    u16* Vtb  = Kbf + NX;
    u16* Ob   = Vtb + NX;   // total 64 MB of workspace

    cvt_bf16<<<1024, 256, 0, stream>>>(query, Xq, (int)NX);
    cvt_bf16<<<1024, 256, 0, stream>>>(key,   Xk, (int)NX);
    cvt_bf16<<<1024, 256, 0, stream>>>(value, Xv, (int)NX);
    cvt_bf16<<<512,  256, 0, stream>>>(Wq,  Wqb,  (int)NW);
    cvt_bf16<<<512,  256, 0, stream>>>(Wk,  Wkb,  (int)NW);
    cvt_bf16<<<512,  256, 0, stream>>>(Wv,  Wvb,  (int)NW);
    cvt_bf16<<<512,  256, 0, stream>>>(Wfc, Wfcb, (int)NW);

    // (M/16)*(N/64) = 4096 strips / 8 waves per block = 512 blocks
    gemm_wmma_nt<<<512, 256, 0, stream>>>(Xq, Wqb, bq, nullptr, Qb,
                                          4096, 1024, 1024, 1);
    gemm_wmma_nt<<<512, 256, 0, stream>>>(Xk, Wkb, bk, nullptr, Kbf,
                                          4096, 1024, 1024, 1);
    gemm_wmma_nt<<<512, 256, 0, stream>>>(Xv, Wvb, bv, nullptr, Vtb,
                                          4096, 1024, 1024, 2);

    float* out  = (float*)d_out;
    float* attn = out + NX;     // tuple order: (output, attention)

    attn_sparsemax<<<2 * 16 * (SEQ / 16), 256, 0, stream>>>(Qb, Kbf, Vtb, attn, Ob);

    gemm_wmma_nt<<<512, 256, 0, stream>>>(Ob, Wfcb, bfc, out, nullptr,
                                          4096, 1024, 1024, 0);
}
